// MultiHeadAttn_54941221651326
// MI455X (gfx1250) — compile-verified, matrix operand order checked by reading, not run on hardware
//
#include <hip/hip_runtime.h>
#include <hip/hip_fp16.h>

typedef _Float16 half_t;
typedef __attribute__((ext_vector_type(16))) _Float16 v16h;
typedef __attribute__((ext_vector_type(8)))  _Float16 v8h;
typedef __attribute__((ext_vector_type(8)))  float    v8f;
typedef __attribute__((ext_vector_type(4)))  int      v4i;

#define B_DIM 4
#define T_DIM 2048
#define C_DIM 1024
#define H_DIM 16
#define D_DIM 64

// ---- CDNA5 async global->LDS copy (guarded: falls back to sync staging) ----
#if defined(__has_builtin)
#if __has_builtin(__builtin_amdgcn_global_load_async_to_lds_b128) && \
    __has_builtin(__builtin_amdgcn_s_wait_asynccnt)
#define HAVE_ASYNC_COPY 1
#endif
#endif

#if defined(HAVE_ASYNC_COPY)
// Generic->AS1/AS3 conversions via integer round-trip: flat global addresses
// equal AS1 addresses; low 32 bits of a generic LDS pointer are the AS3 offset.
// Builtin expects v4i (int4) pointee per compiler diagnostic.
__device__ __forceinline__ void async_copy_b128(const void* g, void* l) {
    __builtin_amdgcn_global_load_async_to_lds_b128(
        (__attribute__((address_space(1))) v4i*)(uintptr_t)g,
        (__attribute__((address_space(3))) v4i*)(uint32_t)(uintptr_t)l,
        0, 0);
}
__device__ __forceinline__ void async_wait0() {
    __builtin_amdgcn_s_wait_asynccnt(0);
}
#else
__device__ __forceinline__ void async_wait0() {}
#endif

// Load a 16-half WMMA fragment (A or B operand) from a rows-contiguous source:
// elements 0..7 from p[0..7], elements 8..15 from p[16..23].
// Caller passes p = rowBase + c0 with c0 = (laneHi ? 8 : 0) [+ 32*kslice],
// which reproduces the ISA's 16-bit A-matrix lane layout (and its mirror for B).
__device__ __forceinline__ v16h load_frag(const half_t* p) {
    v8h lo = *(const v8h*)p;
    v8h hh = *(const v8h*)(p + 16);
    v16h r;
#pragma unroll
    for (int i = 0; i < 8; ++i) { r[i] = lo[i]; r[i + 8] = hh[i]; }
    return r;
}

// ---------------------------------------------------------------------------
// fp32 -> f16 conversion (one-shot pass for weight matrices)
// ---------------------------------------------------------------------------
__global__ __launch_bounds__(256) void f32_to_f16_kernel(
    const float* __restrict__ src, half_t* __restrict__ dst, int n)
{
    const int i = (blockIdx.x * 256 + threadIdx.x) * 4;
    if (i + 3 < n) {
        float4 f = *(const float4*)(src + i);
        dst[i + 0] = (half_t)f.x; dst[i + 1] = (half_t)f.y;
        dst[i + 2] = (half_t)f.z; dst[i + 3] = (half_t)f.w;
    }
}

// ---------------------------------------------------------------------------
// Tiled GEMM: Y[M,N] = X[M,K] @ Wh[N,K]^T + bias[N]   (Wh already f16)
// BM=128, BN=128, BK=32. 256 threads = 8 waves; wave grid 2(M)x4(N),
// each wave computes 64x32 via 4x2 tiles of v_wmma_f32_16x16x32_f16.
// Double-buffered: async copies for tile k+1 overlap WMMAs on tile k.
// ---------------------------------------------------------------------------
template <bool A_F32, bool OUT_F32>
__global__ __launch_bounds__(256) void gemm_xwt_kernel(
    const void* __restrict__ Xv, const half_t* __restrict__ Wh,
    const float* __restrict__ bias, void* __restrict__ Yv,
    int M, int N, int K)
{
    constexpr int LDA = 40; // halves per LDS row (80B, multiple of 16B)
    __shared__ alignas(16) half_t As[2][128 * LDA];
    __shared__ alignas(16) half_t Bs[2][128 * LDA];

    const int tid  = threadIdx.x;
    const int wave = tid >> 5;
    const int lane = tid & 31;
    const int lr   = lane & 15;
    const int hi   = lane >> 4;

    const int blockM = blockIdx.y * 128;
    const int blockN = blockIdx.x * 128;
    const int waveM  = (wave >> 2) * 64;  // 0 or 64
    const int waveN  = (wave & 3) * 32;   // 0,32,64,96

    const int ldRow = tid >> 1;          // 0..127
    const int ldCol = (tid & 1) * 16;    // 0 or 16

    auto stage = [&](int ki, int buf) {
        const int k0 = ki * 32;
        // X tile
        half_t* dstA = &As[buf][ldRow * LDA + ldCol];
        if (A_F32) {
            const float* src = (const float*)Xv + (size_t)(blockM + ldRow) * K + k0 + ldCol;
#pragma unroll
            for (int i = 0; i < 16; i += 4) {
                float4 f = *(const float4*)(src + i);
                dstA[i + 0] = (half_t)f.x; dstA[i + 1] = (half_t)f.y;
                dstA[i + 2] = (half_t)f.z; dstA[i + 3] = (half_t)f.w;
            }
        } else {
            const half_t* src = (const half_t*)Xv + (size_t)(blockM + ldRow) * K + k0 + ldCol;
#if defined(HAVE_ASYNC_COPY)
            async_copy_b128(src, dstA);
            async_copy_b128(src + 8, dstA + 8);
#else
            *(v8h*)(dstA)     = *(const v8h*)(src);
            *(v8h*)(dstA + 8) = *(const v8h*)(src + 8);
#endif
        }
        // W tile (rows of Wh are the Wt columns we need)
        const half_t* srcW = Wh + (size_t)(blockN + ldRow) * K + k0 + ldCol;
        half_t* dstB = &Bs[buf][ldRow * LDA + ldCol];
#if defined(HAVE_ASYNC_COPY)
        async_copy_b128(srcW, dstB);
        async_copy_b128(srcW + 8, dstB + 8);
#else
        *(v8h*)(dstB)     = *(const v8h*)(srcW);
        *(v8h*)(dstB + 8) = *(const v8h*)(srcW + 8);
#endif
    };

    v8f vzero = {};
    v8f acc[4][2];
#pragma unroll
    for (int mt = 0; mt < 4; ++mt)
#pragma unroll
        for (int nt = 0; nt < 2; ++nt) acc[mt][nt] = vzero;

    const int nk = K >> 5;
    stage(0, 0);
    async_wait0();
    __syncthreads();

    for (int ki = 0; ki < nk; ++ki) {
        const int cur = ki & 1;
        if (ki + 1 < nk) stage(ki + 1, cur ^ 1); // overlap with WMMAs below

        v16h bfr[2];
#pragma unroll
        for (int nt = 0; nt < 2; ++nt)
            bfr[nt] = load_frag(&Bs[cur][(waveN + nt * 16 + lr) * LDA + hi * 8]);
#pragma unroll
        for (int mt = 0; mt < 4; ++mt) {
            v16h a = load_frag(&As[cur][(waveM + mt * 16 + lr) * LDA + hi * 8]);
#pragma unroll
            for (int nt = 0; nt < 2; ++nt)
                acc[mt][nt] = __builtin_amdgcn_wmma_f32_16x16x32_f16(
                    false, a, false, bfr[nt], (short)0, acc[mt][nt], false, false);
        }

        async_wait0();
        __syncthreads();
    }

    // ---- epilogue: bias + store (C layout: elem r -> M=r+8*hi, N=lr) ----
#pragma unroll
    for (int nt = 0; nt < 2; ++nt) {
        const int col = blockN + waveN + nt * 16 + lr;
        const float bval = bias[col];
#pragma unroll
        for (int mt = 0; mt < 4; ++mt) {
#pragma unroll
            for (int r = 0; r < 8; ++r) {
                const int row = blockM + waveM + mt * 16 + r + 8 * hi;
                const float v = acc[mt][nt][r] + bval;
                if (OUT_F32) ((float*)Yv)[(size_t)row * N + col] = v;
                else         ((half_t*)Yv)[(size_t)row * N + col] = (half_t)v;
            }
        }
    }
}

// ---------------------------------------------------------------------------
// Flash attention: one workgroup = 128 query rows of one (b,h); 8 waves,
// each wave owns 16 query rows. Key/value blocks of 64 staged through LDS,
// double-buffered: async K copies + V transpose for block kb+1 overlap the
// S/softmax/PV compute on block kb. V is transposed on store so every PV
// B-fragment read stays a contiguous ds_load_b128.
// ---------------------------------------------------------------------------
__global__ __launch_bounds__(256) void flash_attn_kernel(
    const half_t* __restrict__ qh, const half_t* __restrict__ kh,
    const half_t* __restrict__ vh, half_t* __restrict__ oh)
{
    constexpr int LDK = 72; // halves per LDS row (144B, multiple of 16B)
    __shared__ alignas(16) half_t Ks[2][64 * LDK];    // K rows: [key][d]
    __shared__ alignas(16) half_t Vt[2][64 * LDK];    // V transposed: [d][key]
    __shared__ alignas(16) half_t Ps[8][16 * LDK];    // per-wave P staging

    const int tid  = threadIdx.x;
    const int wave = tid >> 5;
    const int lane = tid & 31;
    const int lr   = lane & 15;
    const int hi   = lane >> 4;

    const int bh = blockIdx.y;
    const int b  = bh >> 4;
    const int h  = bh & 15;
    const int qbase = blockIdx.x * 128;
    const int wq    = qbase + wave * 16;

    const size_t headOff = ((size_t)b * T_DIM) * C_DIM + (size_t)h * D_DIM;

    const int key  = tid >> 2;        // 0..63  (cooperative staging coords)
    const int dseg = (tid & 3) * 16;  // 0,16,32,48

    auto stageKV = [&](int kb, int buf) {
        const size_t gOff = headOff + (size_t)(kb * 64 + key) * C_DIM + dseg;
        half_t* kdst = &Ks[buf][key * LDK + dseg];
#if defined(HAVE_ASYNC_COPY)
        async_copy_b128(kh + gOff, kdst);
        async_copy_b128(kh + gOff + 8, kdst + 8);
#else
        *(v8h*)(kdst)     = *(const v8h*)(kh + gOff);
        *(v8h*)(kdst + 8) = *(const v8h*)(kh + gOff + 8);
#endif
        v8h v0 = *(const v8h*)(vh + gOff);
        v8h v1 = *(const v8h*)(vh + gOff + 8);
#pragma unroll
        for (int i = 0; i < 8; ++i) {
            Vt[buf][(dseg + i) * LDK + key]     = v0[i];
            Vt[buf][(dseg + 8 + i) * LDK + key] = v1[i];
        }
    };

    // Q fragments for this wave's 16 rows (kept in registers the whole time)
    v16h qf[2];
    {
        const half_t* qrow = qh + headOff + (size_t)(wq + lr) * C_DIM;
#pragma unroll
        for (int s = 0; s < 2; ++s) qf[s] = load_frag(qrow + s * 32 + hi * 8);
    }

    float mrow[8], lrow[8];
    v8f vzero = {};
    v8f oacc[4];
#pragma unroll
    for (int r = 0; r < 8; ++r) { mrow[r] = -1e30f; lrow[r] = 0.0f; }
#pragma unroll
    for (int dt = 0; dt < 4; ++dt) oacc[dt] = vzero;

    const float scale = 0.125f; // 1/sqrt(64)
    const int nkb = (qbase + 128) >> 6; // causal: key blocks up to query block end

    stageKV(0, 0);
    async_wait0();
    __syncthreads();

    for (int kb = 0; kb < nkb; ++kb) {
        const int cur = kb & 1;
        if (kb + 1 < nkb) stageKV(kb + 1, cur ^ 1); // overlap with compute below
        if (kb + 2 < nkb) { // prefetch two blocks ahead into cache
            const size_t pOff = headOff + (size_t)((kb + 2) * 64 + key) * C_DIM + dseg;
            __builtin_prefetch(kh + pOff, 0, 3);
            __builtin_prefetch(vh + pOff, 0, 3);
        }

        // ---- S = Q K^T (4 n-tiles of 16 keys, K=64 = 2 WMMA slices) ----
        v8f sacc[4];
#pragma unroll
        for (int nt = 0; nt < 4; ++nt) {
            sacc[nt] = vzero;
#pragma unroll
            for (int s = 0; s < 2; ++s) {
                v16h bf = load_frag(&Ks[cur][(nt * 16 + lr) * LDK + s * 32 + hi * 8]);
                sacc[nt] = __builtin_amdgcn_wmma_f32_16x16x32_f16(
                    false, qf[s], false, bf, (short)0, sacc[nt], false, false);
            }
        }

        // ---- scale + causal mask + per-row max ----
        float mnew[8];
#pragma unroll
        for (int r = 0; r < 8; ++r) mnew[r] = -1e30f;
#pragma unroll
        for (int nt = 0; nt < 4; ++nt) {
            const int kcol = kb * 64 + nt * 16 + lr;
#pragma unroll
            for (int r = 0; r < 8; ++r) {
                const int row = wq + r + 8 * hi;
                float v = sacc[nt][r] * scale;
                v = (kcol <= row) ? v : -1e30f;
                sacc[nt][r] = v;
                mnew[r] = fmaxf(mnew[r], v);
            }
        }
#pragma unroll
        for (int off = 1; off < 16; off <<= 1)
#pragma unroll
            for (int r = 0; r < 8; ++r)
                mnew[r] = fmaxf(mnew[r], __shfl_xor(mnew[r], off, 32));

        float alpha[8];
#pragma unroll
        for (int r = 0; r < 8; ++r) {
            const float mn = fmaxf(mrow[r], mnew[r]);
            alpha[r] = __expf(mrow[r] - mn);
            mrow[r] = mn;
        }

        // ---- P = exp(S - m), row sums, stage P into LDS (C-layout -> A-layout) ----
        float lsum[8];
#pragma unroll
        for (int r = 0; r < 8; ++r) lsum[r] = 0.0f;
#pragma unroll
        for (int nt = 0; nt < 4; ++nt) {
#pragma unroll
            for (int r = 0; r < 8; ++r) {
                const float p = __expf(sacc[nt][r] - mrow[r]);
                lsum[r] += p;
                Ps[wave][(r + 8 * hi) * LDK + nt * 16 + lr] = (half_t)p;
            }
        }
#pragma unroll
        for (int off = 1; off < 16; off <<= 1)
#pragma unroll
            for (int r = 0; r < 8; ++r)
                lsum[r] += __shfl_xor(lsum[r], off, 32);
#pragma unroll
        for (int r = 0; r < 8; ++r) lrow[r] = lrow[r] * alpha[r] + lsum[r];

        // ---- rescale O, then O += P V ----
#pragma unroll
        for (int dt = 0; dt < 4; ++dt)
#pragma unroll
            for (int r = 0; r < 8; ++r) oacc[dt][r] *= alpha[r];

#pragma unroll
        for (int s = 0; s < 2; ++s) {
            v16h pa = load_frag(&Ps[wave][lr * LDK + s * 32 + hi * 8]);
#pragma unroll
            for (int dt = 0; dt < 4; ++dt) {
                v16h vb = load_frag(&Vt[cur][(dt * 16 + lr) * LDK + s * 32 + hi * 8]);
                oacc[dt] = __builtin_amdgcn_wmma_f32_16x16x32_f16(
                    false, pa, false, vb, (short)0, oacc[dt], false, false);
            }
        }

        async_wait0();
        __syncthreads();
    }

    // ---- normalize and write (f16, [B][T][H*D] layout) ----
#pragma unroll
    for (int r = 0; r < 8; ++r) {
        const float inv = 1.0f / lrow[r];
        const int row = wq + r + 8 * hi;
        half_t* orow = oh + headOff + (size_t)row * C_DIM;
#pragma unroll
        for (int dt = 0; dt < 4; ++dt)
            orow[dt * 16 + lr] = (half_t)(oacc[dt][r] * inv);
    }
}

// ---------------------------------------------------------------------------
extern "C" void kernel_launch(void* const* d_in, const int* in_sizes, int n_in,
                              void* d_out, int out_size, void* d_ws, size_t ws_size,
                              hipStream_t stream) {
    (void)in_sizes; (void)n_in; (void)out_size; (void)ws_size;

    const float* query = (const float*)d_in[0];
    const float* key   = (const float*)d_in[1];
    const float* value = (const float*)d_in[2];
    // d_in[3] = positions (unused, rope pass-through), d_in[4] = mask (causal, recomputed)
    const float* Wq = (const float*)d_in[5];
    const float* bq = (const float*)d_in[6];
    const float* Wk = (const float*)d_in[7];
    const float* bk = (const float*)d_in[8];
    const float* Wv = (const float*)d_in[9];
    const float* bv = (const float*)d_in[10];
    const float* Wo = (const float*)d_in[11];
    const float* bo = (const float*)d_in[12];

    const size_t elems = (size_t)B_DIM * T_DIM * C_DIM; // 8M elements, 16MB as f16
    const size_t welems = (size_t)C_DIM * C_DIM;        // 1M elements, 2MB as f16
    half_t* qws  = (half_t*)d_ws;
    half_t* kws  = qws + elems;
    half_t* vws  = kws + elems;
    half_t* aws  = vws + elems;
    half_t* wqh  = aws + elems;
    half_t* wkh  = wqh + welems;
    half_t* wvh  = wkh + welems;
    half_t* woh  = wvh + welems;

    // one-shot fp32 -> f16 weight conversion (enables async LDS staging in GEMMs)
    {
        const int n = (int)welems;
        const int grid = n / (256 * 4);
        f32_to_f16_kernel<<<grid, 256, 0, stream>>>(Wq, wqh, n);
        f32_to_f16_kernel<<<grid, 256, 0, stream>>>(Wk, wkh, n);
        f32_to_f16_kernel<<<grid, 256, 0, stream>>>(Wv, wvh, n);
        f32_to_f16_kernel<<<grid, 256, 0, stream>>>(Wo, woh, n);
    }

    const int M = B_DIM * T_DIM, N = C_DIM, K = C_DIM;
    dim3 gblk(N / 128, M / 128); // (8, 64)
    gemm_xwt_kernel<true, false><<<gblk, 256, 0, stream>>>(query, wqh, bq, qws, M, N, K);
    gemm_xwt_kernel<true, false><<<gblk, 256, 0, stream>>>(key,   wkh, bk, kws, M, N, K);
    gemm_xwt_kernel<true, false><<<gblk, 256, 0, stream>>>(value, wvh, bv, vws, M, N, K);

    dim3 ablk(T_DIM / 128, B_DIM * H_DIM); // (16, 64)
    flash_attn_kernel<<<ablk, 256, 0, stream>>>(qws, kws, vws, aws);

    gemm_xwt_kernel<false, true><<<gblk, 256, 0, stream>>>(aws, woh, bo, d_out, M, N, K);
}
